// RV2BEVFrustumAttn_59133109732255
// MI455X (gfx1250) — matrix operand order repaired and checked by reading, not compiled, not run on hardware
//
#include <hip/hip_runtime.h>
#include <hip/hip_bf16.h>
#include <math.h>

typedef __attribute__((ext_vector_type(16))) _Float16 v16h;
typedef __attribute__((ext_vector_type(8)))  _Float16 v8h;
typedef __attribute__((ext_vector_type(8)))  float    v8f;
typedef _Float16 f16;

#define B_   4
#define HRV_ 32
#define WRV_ 512
#define NPIX_RV 16384      // 32*512
#define HB_  256
#define WB_  256
#define NPIX_BEV 65536     // 256*256
#define D_   128

constexpr int AK_F32_PIX = 0, AK_F32_PLANAR = 1, AK_F16_PIX = 2;
constexpr int OK_F16 = 0, OK_F32 = 1, OK_PLANAR = 2;

__device__ __constant__ float c_elev[32] = {
  -30.67f,-29.33f,-28.0f,-26.66f,-25.33f,-24.0f,-22.67f,-21.33f,-20.0f,-18.67f,
  -17.33f,-16.0f,-14.67f,-13.33f,-12.0f,-10.67f,-9.33f,-8.0f,-6.66f,-5.33f,
  -4.0f,-2.67f,-1.33f,0.0f,1.33f,2.67f,4.0f,5.33f,6.67f,8.0f,9.33f,10.67f };

__device__ __forceinline__ float gelu_f(float x) {
  return 0.5f * x * (1.0f + erff(x * 0.70710678118654752f));
}
__device__ __forceinline__ float sigm_f(float x) {
  return 1.0f / (1.0f + expf(-x));
}

// ---------------------------------------------------------------------------
// Generic WMMA GEMM.  Out[b,pix,0:COUT] = act( A[b,pix,:]*Wt + bias + extras ).
// Wt is f16 K-major: Wt[k*COUT+n].  Block = 256 thr (8 waves), 256 px/block,
// each wave owns a 32-px strip (2 M-tiles) x full COUT (2 WMMAs per B-frag).
// ---------------------------------------------------------------------------
template<int CIN, int COUT, int AK, int OK, bool GELU, int EXTRA>
__global__ __launch_bounds__(256) void gemm_wmma(
    const void* __restrict__ Ap, const f16* __restrict__ Wt,
    const float* __restrict__ bias,
    const float* __restrict__ exX, const float* __restrict__ exW, long exBS,
    void* __restrict__ Op, int npix, int ldA, int ldOut)
{
  constexpr int NT    = COUT / 16;
  constexpr int KS    = CIN / 32;
  constexpr int LDS_A = 40;          // 256x32 f16 A-tile, padded (80B rows)
  constexpr int LDW   = COUT + 8;    // 32xCOUT f16 W-tile (16B-multiple rows)
  constexpr int WCH   = (32 * COUT) / 8;   // 8-half chunks in W tile
  __shared__ __align__(16) f16 sA[256 * LDS_A];
  __shared__ __align__(16) f16 sW[32 * LDW];

  const int t    = threadIdx.x;
  const int wv   = t >> 5;
  const int lane = t & 31;
  const int b    = blockIdx.y;
  const long pix0 = (long)blockIdx.x * 256;

  v8f acc[2][NT];
  #pragma unroll
  for (int m = 0; m < 2; ++m)
    #pragma unroll
    for (int i = 0; i < NT; ++i)
      #pragma unroll
      for (int j = 0; j < 8; ++j) acc[m][i][j] = 0.0f;

  for (int ks = 0; ks < KS; ++ks) {
    __syncthreads();
    // ---- stage A tile (256 px x 32 k) as f16, one pixel per thread (planar: one k x 32 px per thread) ----
    if constexpr (AK == AK_F32_PLANAR) {
      const float* Ab = (const float*)Ap + (long)b * CIN * npix;
      int k = t >> 3, c0 = (t & 7) * 32;
      const float4* s4 = (const float4*)(Ab + (long)(ks * 32 + k) * npix + pix0 + c0);
      #pragma unroll
      for (int j = 0; j < 8; ++j) {
        float4 q = s4[j];
        sA[(c0 + 4 * j + 0) * LDS_A + k] = (f16)q.x;
        sA[(c0 + 4 * j + 1) * LDS_A + k] = (f16)q.y;
        sA[(c0 + 4 * j + 2) * LDS_A + k] = (f16)q.z;
        sA[(c0 + 4 * j + 3) * LDS_A + k] = (f16)q.w;
      }
    } else if constexpr (AK == AK_F32_PIX) {
      const float* src = (const float*)Ap + ((long)b * npix + pix0 + t) * ldA + ks * 32;
      if (ks + 1 < KS) __builtin_prefetch(src + 32, 0, 0);
      const float4* s4 = (const float4*)src;
      v8h* dst = (v8h*)&sA[t * LDS_A];
      #pragma unroll
      for (int j = 0; j < 4; ++j) {
        float4 q0 = s4[2 * j], q1 = s4[2 * j + 1];
        v8h h;
        h[0] = (f16)q0.x; h[1] = (f16)q0.y; h[2] = (f16)q0.z; h[3] = (f16)q0.w;
        h[4] = (f16)q1.x; h[5] = (f16)q1.y; h[6] = (f16)q1.z; h[7] = (f16)q1.w;
        dst[j] = h;
      }
    } else { // f16 pixel-major
      const f16* src = (const f16*)Ap + ((long)b * npix + pix0 + t) * ldA + ks * 32;
      if (ks + 1 < KS) __builtin_prefetch(src + 32, 0, 0);
      const v8h* s8 = (const v8h*)src;
      v8h* dst = (v8h*)&sA[t * LDS_A];
      #pragma unroll
      for (int j = 0; j < 4; ++j) dst[j] = s8[j];
    }
    // ---- stage W tile (32 k x COUT n) in 16B chunks ----
    #pragma unroll
    for (int it = 0; it < (WCH + 255) / 256; ++it) {
      int c = t + it * 256;
      if (c < WCH) {
        int hb = c * 8;
        int k = hb / COUT, n = hb - k * COUT;
        *(v8h*)&sW[k * LDW + n] =
            *(const v8h*)(Wt + (long)(ks * 32 + k) * COUT + n);
      }
    }
    __syncthreads();

    // ---- fragments + WMMA ----
    const f16* ar = &sA[(wv * 32 + (lane & 15)) * LDS_A + ((lane >> 4) << 3)];
    v16h af0, af1;
    #pragma unroll
    for (int i = 0; i < 8; ++i) {
      af0[i] = ar[i];               af0[8 + i] = ar[16 + i];
      af1[i] = ar[16 * LDS_A + i];  af1[8 + i] = ar[16 * LDS_A + 16 + i];
    }
    #pragma unroll
    for (int nt = 0; nt < NT; ++nt) {
      const f16* br = &sW[((lane >> 4) << 3) * LDW + nt * 16 + (lane & 15)];   // n = lane%16, 16 k values (table order)
      v16h bf;
      #pragma unroll
      for (int i = 0; i < 16; ++i) bf[i] = br[(i + (i & 8)) * LDW];
      acc[0][nt] = __builtin_amdgcn_wmma_f32_16x16x32_f16(
          false, af0, false, bf, (short)0, acc[0][nt], false, false);
      acc[1][nt] = __builtin_amdgcn_wmma_f32_16x16x32_f16(
          false, af1, false, bf, (short)0, acc[1][nt], false, false);
    }
  }

  // ---- epilogue ----
  const int mb = (lane >> 4) << 3;
  const int nc = lane & 15;
  float exv[EXTRA > 0 ? 16 * EXTRA : 1];
  if constexpr (EXTRA > 0) {
    #pragma unroll
    for (int m = 0; m < 2; ++m)
      #pragma unroll
      for (int r = 0; r < 8; ++r) {
        long pix = pix0 + wv * 32 + m * 16 + mb + r;
        #pragma unroll
        for (int e = 0; e < EXTRA; ++e)
          exv[(m * 8 + r) * EXTRA + e] = exX[b * exBS + pix * EXTRA + e];
      }
  }
  #pragma unroll
  for (int m = 0; m < 2; ++m) {
    #pragma unroll
    for (int nt = 0; nt < NT; ++nt) {
      int n = nt * 16 + nc;
      float bv = bias[n];
      float ew[EXTRA > 0 ? EXTRA : 1];
      if constexpr (EXTRA > 0) {
        #pragma unroll
        for (int e = 0; e < EXTRA; ++e) ew[e] = exW[e * COUT + n];
      }
      float vals[8];
      #pragma unroll
      for (int r = 0; r < 8; ++r) {
        float val = acc[m][nt][r] + bv;
        if constexpr (EXTRA > 0) {
          #pragma unroll
          for (int e = 0; e < EXTRA; ++e) val += exv[(m * 8 + r) * EXTRA + e] * ew[e];
        }
        if constexpr (GELU) val = gelu_f(val);
        vals[r] = val;
      }
      long pixb = pix0 + wv * 32 + m * 16 + mb;
      if constexpr (OK == OK_F16) {
        f16* o = (f16*)Op;
        #pragma unroll
        for (int r = 0; r < 8; ++r)
          o[((long)b * npix + pixb + r) * ldOut + n] = (f16)vals[r];
      } else if constexpr (OK == OK_F32) {
        float* o = (float*)Op;
        #pragma unroll
        for (int r = 0; r < 8; ++r)
          o[((long)b * npix + pixb + r) * ldOut + n] = vals[r];
      } else { // planar (B, COUT, H, W): 8 consecutive pixels -> 2x float4
        float* o = (float*)Op + (long)b * COUT * npix + (long)n * npix + pixb;
        *(float4*)o       = make_float4(vals[0], vals[1], vals[2], vals[3]);
        *(float4*)(o + 4) = make_float4(vals[4], vals[5], vals[6], vals[7]);
      }
    }
  }
}

// ---------------------------------------------------------------------------
// 3x3 wrap-padded conv 128->64 as 9 accumulating WMMA GEMMs over f16 input.
// 256 px per block (one row strip), 2 M-tiles per wave.
// ---------------------------------------------------------------------------
__global__ __launch_bounds__(256) void conv3_wmma(
    const f16* __restrict__ h, const f16* __restrict__ Wt, float* __restrict__ out)
{
  constexpr int LDS_A = 40, LDW = 72;
  __shared__ __align__(16) f16 sA[256 * LDS_A];
  __shared__ __align__(16) f16 sW[32 * LDW];
  const int t = threadIdx.x, wv = t >> 5, lane = t & 31;
  const int w0 = blockIdx.x * 256, hrow = blockIdx.y, b = blockIdx.z;

  v8f acc[2][4];
  #pragma unroll
  for (int m = 0; m < 2; ++m)
    #pragma unroll
    for (int i = 0; i < 4; ++i)
      #pragma unroll
      for (int j = 0; j < 8; ++j) acc[m][i][j] = 0.0f;

  for (int tap = 0; tap < 9; ++tap) {
    int dy = tap / 3 - 1, dx = tap % 3 - 1;
    int hh = (hrow + dy + 32) & 31;
    for (int ks = 0; ks < 4; ++ks) {
      __syncthreads();
      {
        int ww = (w0 + t + dx + 512) & 511;
        const v8h* s8 = (const v8h*)(h + (((long)b * 32 + hh) * 512 + ww) * 128 + ks * 32);
        v8h* dst = (v8h*)&sA[t * LDS_A];
        #pragma unroll
        for (int j = 0; j < 4; ++j) dst[j] = s8[j];
      }
      { // 32x64 halfs = 256 chunks of 8: one per thread
        int hb = t * 8;
        int k = hb >> 6, n = hb & 63;
        *(v8h*)&sW[k * LDW + n] =
            *(const v8h*)(Wt + ((long)(tap * 128) + ks * 32 + k) * 64 + n);
      }
      __syncthreads();
      const f16* ar = &sA[(wv * 32 + (lane & 15)) * LDS_A + ((lane >> 4) << 3)];
      v16h af0, af1;
      #pragma unroll
      for (int i = 0; i < 8; ++i) {
        af0[i] = ar[i];               af0[8 + i] = ar[16 + i];
        af1[i] = ar[16 * LDS_A + i];  af1[8 + i] = ar[16 * LDS_A + 16 + i];
      }
      #pragma unroll
      for (int nt = 0; nt < 4; ++nt) {
        const f16* br = &sW[((lane >> 4) << 3) * LDW + nt * 16 + (lane & 15)];
        v16h bf;
        #pragma unroll
        for (int i = 0; i < 16; ++i) bf[i] = br[(i + (i & 8)) * LDW];
        acc[0][nt] = __builtin_amdgcn_wmma_f32_16x16x32_f16(
            false, af0, false, bf, (short)0, acc[0][nt], false, false);
        acc[1][nt] = __builtin_amdgcn_wmma_f32_16x16x32_f16(
            false, af1, false, bf, (short)0, acc[1][nt], false, false);
      }
    }
  }
  const int mb = (lane >> 4) << 3, nc = lane & 15;
  #pragma unroll
  for (int m = 0; m < 2; ++m)
    #pragma unroll
    for (int nt = 0; nt < 4; ++nt)
      #pragma unroll
      for (int r = 0; r < 8; ++r) {
        long pix = (long)hrow * 512 + w0 + wv * 32 + m * 16 + mb + r;
        out[((long)b * NPIX_RV + pix) * 64 + nt * 16 + nc] = acc[m][nt][r];
      }
}

// ---------------------------------------------------------------------------
// GroupNorm: one block per (b,g); produces per-channel scale/shift.
// ---------------------------------------------------------------------------
__global__ void gn_reduce(const float* __restrict__ x, const float* __restrict__ gw,
                          const float* __restrict__ gb, float* scale, float* shift,
                          int C, int gcShift, int npix)
{
  int GC = 1 << gcShift;
  int G = C >> gcShift;
  int b = blockIdx.x / G, g = blockIdx.x % G;
  const float* base = x + (long)b * npix * C + g * GC;
  float s = 0.f, s2 = 0.f;
  int total = npix << gcShift;
  for (int i = threadIdx.x; i < total; i += 256) {
    int p = i >> gcShift, c = i & (GC - 1);
    float v = base[(long)p * C + c];
    s += v; s2 += v * v;
  }
  __shared__ float sh[256], sh2[256];
  sh[threadIdx.x] = s; sh2[threadIdx.x] = s2;
  __syncthreads();
  for (int st = 128; st > 0; st >>= 1) {
    if (threadIdx.x < st) { sh[threadIdx.x] += sh[threadIdx.x + st];
                            sh2[threadIdx.x] += sh2[threadIdx.x + st]; }
    __syncthreads();
  }
  if (threadIdx.x < GC) {
    float n = (float)total;
    float m = sh[0] / n, var = sh2[0] / n - m * m;
    float inv = rsqrtf(var + 1e-5f);
    int c = g * GC + threadIdx.x;
    float sc = gw[c] * inv;
    scale[b * C + c] = sc;
    shift[b * C + c] = gb[c] - m * sc;
  }
}

__global__ void gn_apply_gelu(const float* __restrict__ x, const float* __restrict__ sc,
                              const float* __restrict__ sf, f16* __restrict__ o,
                              int C, int npix)
{
  long i = (long)blockIdx.x * 256 + threadIdx.x;
  int c = (int)(i % C);
  int b = (int)(i / ((long)npix * C));
  float v = x[i] * sc[b * C + c] + sf[b * C + c];
  o[i] = (f16)gelu_f(v);
}

// ---------------------------------------------------------------------------
// Head: GN2-apply + gelu + rh3(64->2) + mu/sigma + ref points + sig_feat.
// ---------------------------------------------------------------------------
__global__ void head_kernel(const float* __restrict__ h2, const float* __restrict__ sc,
                            const float* __restrict__ sf, const float* __restrict__ rh3w,
                            const float* __restrict__ rh3b, const float* __restrict__ geom,
                            const float* __restrict__ L, float* __restrict__ muo,
                            float* __restrict__ sgo, float* __restrict__ refb,
                            float* __restrict__ sigf)
{
  int i = blockIdx.x * 256 + threadIdx.x;           // b*16384 + q
  int b = i >> 14, q = i & 16383;
  const float* hp = h2 + (long)i * 64;
  const float* scb = sc + b * 64;
  const float* sfb = sf + b * 64;
  float o0 = rh3b[0], o1 = rh3b[1];
  for (int c = 0; c < 64; ++c) {
    float g = gelu_f(hp[c] * scb[c] + sfb[c]);
    o0 += g * rh3w[c];
    o1 += g * rh3w[64 + c];
  }
  float mus = sigm_f(o0);
  float mu  = mus * 55.0f;
  float sg  = fmaxf(fmaxf(o1, 0.0f), 1e-3f) * 55.0f;
  float ux = geom[q * 3], uy = geom[q * 3 + 1], uz = geom[q * 3 + 2];
  float px = mu * ux, py = mu * uy, pz = mu * uz;
  float ex = px * L[0] + py * L[4] + pz * L[8]  + L[12];
  float ey = px * L[1] + py * L[5] + pz * L[9]  + L[13];
  float rx = sigm_f(4.0f * ((ex + 55.0f) / 110.0f - 0.5f));
  float ry = sigm_f(4.0f * ((ey + 55.0f) / 110.0f - 0.5f));
  muo[i] = mus;            // mu / RMAX == sigmoid(o0)
  sgo[i] = sg;
  refb[(long)i * 2]     = rx;
  refb[(long)i * 2 + 1] = ry;
  sigf[(long)i * 2]     = sg;
  sigf[(long)i * 2 + 1] = 1.0f / (sg + 1e-6f);
}

// ---------------------------------------------------------------------------
// MSDA sampling: per (b,q,head): softmax(6) + 6x bilinear gather of 16-ch f16.
// ---------------------------------------------------------------------------
__global__ void msda_sample(const float* __restrict__ P, const float* __restrict__ refb,
                            const f16* __restrict__ v, f16* __restrict__ cat)
{
  int i = blockIdx.x * 256 + threadIdx.x;  // < B*16384*8
  int hd = i & 7, bq = i >> 3;
  int b = bq >> 14;
  const float* Pp = P + (long)bq * 144;
  float rx = refb[(long)bq * 2], ry = refb[(long)bq * 2 + 1];
  float aw[6]; float mx = -1e30f;
  #pragma unroll
  for (int p = 0; p < 6; ++p) { aw[p] = Pp[96 + hd * 6 + p]; mx = fmaxf(mx, aw[p]); }
  float ssum = 0.f;
  #pragma unroll
  for (int p = 0; p < 6; ++p) { aw[p] = expf(aw[p] - mx); ssum += aw[p]; }
  float inv = 1.0f / ssum;

  float acc[16];
  #pragma unroll
  for (int c = 0; c < 16; ++c) acc[c] = 0.f;
  const f16* vb = v + (long)b * NPIX_BEV * 128 + hd * 16;

  for (int p = 0; p < 6; ++p) {
    float ox = Pp[hd * 12 + p * 2], oy = Pp[hd * 12 + p * 2 + 1];
    float xs = rx * 256.0f + ox - 0.5f;
    float ys = ry * 256.0f + oy - 0.5f;
    float x0 = floorf(xs), y0 = floorf(ys);
    float wx = xs - x0, wy = ys - y0;
    int ix = (int)x0, iy = (int)y0;
    float a = aw[p] * inv;
    float cw[4] = { (1.f - wx) * (1.f - wy) * a, wx * (1.f - wy) * a,
                    (1.f - wx) * wy * a,         wx * wy * a };
    #pragma unroll
    for (int cy = 0; cy < 2; ++cy)
      #pragma unroll
      for (int cx = 0; cx < 2; ++cx) {
        int X = ix + cx, Y = iy + cy;
        if (X >= 0 && X < 256 && Y >= 0 && Y < 256) {
          const v8h* s8 = (const v8h*)(vb + (long)(Y * 256 + X) * 128);
          v8h a0 = s8[0], a1 = s8[1];
          float w = cw[cy * 2 + cx];
          #pragma unroll
          for (int c = 0; c < 8; ++c) {
            acc[c]     += w * (float)a0[c];
            acc[8 + c] += w * (float)a1[c];
          }
        }
      }
  }
  v8h o0, o1;
  #pragma unroll
  for (int c = 0; c < 8; ++c) { o0[c] = (f16)acc[c]; o1[c] = (f16)acc[8 + c]; }
  v8h* o = (v8h*)(cat + (long)bq * 256 + 128 + hd * 16);
  o[0] = o0; o[1] = o1;
}

// ---------------------------------------------------------------------------
// Prep kernels: geometry, weight transposes/folds to f16 K-major.
// ---------------------------------------------------------------------------
__global__ void k_geom(float* g) {
  int t = blockIdx.x * 256 + threadIdx.x;
  if (t >= NPIX_RV) return;
  int h = t >> 9, w = t & 511;
  float az = -3.14159265358979f + (float)w * (6.28318530717959f / 512.0f);
  float el = c_elev[31 - h] * (3.14159265358979f / 180.0f);
  float sa, ca; sincosf(az, &sa, &ca);
  float ce = cosf(el), se = sinf(el);
  g[t * 3 + 0] = ca * ce; g[t * 3 + 1] = sa * ce; g[t * 3 + 2] = se;
}

// Wt[k*ldo+n] = sum_c A[n*ldA+c] * Bm[c*ldB+k]   (128x128x128 fold)
__global__ void k_fold(const float* __restrict__ Am, int ldA,
                       const float* __restrict__ Bm, int ldB,
                       f16* __restrict__ Wt, int ldo)
{
  int i = blockIdx.x * 256 + threadIdx.x;
  if (i >= 128 * 128) return;
  int k = i >> 7, n = i & 127;
  float s = 0.f;
  for (int c = 0; c < 128; ++c) s += Am[n * ldA + c] * Bm[c * ldB + k];
  Wt[k * ldo + n] = (f16)s;
}

// Wt[k*ldo+colOff+n] = W[n*ldW+k]
__global__ void k_convT(const float* __restrict__ W, int ldW, int K, int Cout,
                        f16* __restrict__ Wt, int ldo, int colOff)
{
  int i = blockIdx.x * 256 + threadIdx.x;
  if (i >= K * Cout) return;
  int k = i / Cout, n = i - k * Cout;
  Wt[k * ldo + colOff + n] = (f16)W[n * ldW + k];
}

__global__ void k_foldb(const float* __restrict__ Am, int ldA,
                        const float* __restrict__ bin, const float* __restrict__ b0,
                        float* __restrict__ bout)
{
  int n = threadIdx.x;
  if (n >= 128) return;
  float s = b0[n];
  for (int c = 0; c < 128; ++c) s += Am[n * ldA + c] * bin[c];
  bout[n] = s;
}

__global__ void k_misc(const float* so_b, const float* aw_b, float* soawB,
                       const float* rh1_w, float* exW3,
                       const float* qs1_w, float* wsig)
{
  int t = blockIdx.x * 256 + threadIdx.x;
  if (t < 144) soawB[t] = (t < 96) ? so_b[t] : aw_b[t - 96];
  if (t < 384) { int e = t / 128, n = t % 128; exW3[e * 128 + n] = rh1_w[n * 131 + 128 + e]; }
  if (t < 256) { int e = t >> 7, n = t & 127;  wsig[e * 128 + n] = qs1_w[n * 130 + 128 + e]; }
}

__global__ void k_rh2(const float* __restrict__ w, f16* __restrict__ o)
{
  int i = blockIdx.x * 256 + threadIdx.x;
  if (i >= 9 * 128 * 64) return;
  int tap = i / (128 * 64); int r = i - tap * 128 * 64;
  int k = r >> 6, n = r & 63;
  o[((tap * 128) + k) * 64 + n] = (f16)w[(n * 128 + k) * 9 + tap];
}

// ---------------------------------------------------------------------------
extern "C" void kernel_launch(void* const* d_in, const int* in_sizes, int n_in,
                              void* d_out, int out_size, void* d_ws, size_t ws_size,
                              hipStream_t stream)
{
  const float* x_rv  = (const float*)d_in[0];
  const float* bev   = (const float*)d_in[1];
  const float* L     = (const float*)d_in[2];
  const float* pq_w  = (const float*)d_in[3];
  const float* pq_b  = (const float*)d_in[4];
  const float* pv_w  = (const float*)d_in[5];
  const float* pv_b  = (const float*)d_in[6];
  const float* po_w  = (const float*)d_in[7];
  const float* po_b  = (const float*)d_in[8];
  const float* qs1_w = (const float*)d_in[9];
  const float* qs1_b = (const float*)d_in[10];
  const float* qs2_w = (const float*)d_in[11];
  const float* qs2_b = (const float*)d_in[12];
  const float* rh1_w = (const float*)d_in[13];
  const float* rh1_b = (const float*)d_in[14];
  const float* gn1_w = (const float*)d_in[15];
  const float* gn1_b = (const float*)d_in[16];
  const float* rh2_w = (const float*)d_in[17];
  const float* gn2_w = (const float*)d_in[18];
  const float* gn2_b = (const float*)d_in[19];
  const float* rh3_w = (const float*)d_in[20];
  const float* rh3_b = (const float*)d_in[21];
  const float* so_w  = (const float*)d_in[22];
  const float* so_b  = (const float*)d_in[23];
  const float* aw_w  = (const float*)d_in[24];
  const float* aw_b  = (const float*)d_in[25];
  const float* vp_w  = (const float*)d_in[26];
  const float* vp_b  = (const float*)d_in[27];
  const float* op_w  = (const float*)d_in[28];
  const float* op_b  = (const float*)d_in[29];

  char* ws = (char*)d_ws;
  size_t off = 0;
  auto alloc = [&](size_t n) { size_t o = off; off += (n + 255) & ~(size_t)255; return o; };

  f16*   v     = (f16*)  (ws + alloc((size_t)B_ * NPIX_BEV * 128 * 2));
  float* hpre  = (float*)(ws + alloc((size_t)B_ * NPIX_RV * 128 * 4));
  f16*   h16   = (f16*)  (ws + alloc((size_t)B_ * NPIX_RV * 128 * 2));
  float* h2pre = (float*)(ws + alloc((size_t)B_ * NPIX_RV * 64 * 4));
  f16*   h1    = (f16*)  (ws + alloc((size_t)B_ * NPIX_RV * 128 * 2));
  f16*   cat   = (f16*)  (ws + alloc((size_t)B_ * NPIX_RV * 256 * 2));
  float* Pb    = (float*)(ws + alloc((size_t)B_ * NPIX_RV * 144 * 4));
  float* refb  = (float*)(ws + alloc((size_t)B_ * NPIX_RV * 2 * 4));
  float* sigf  = (float*)(ws + alloc((size_t)B_ * NPIX_RV * 2 * 4));
  float* geom  = (float*)(ws + alloc((size_t)NPIX_RV * 3 * 4));
  f16* vWt    = (f16*)(ws + alloc(128 * 128 * 2));
  f16* WaT    = (f16*)(ws + alloc(128 * 128 * 2));
  f16* qs2Wt  = (f16*)(ws + alloc(128 * 128 * 2));
  f16* soawWt = (f16*)(ws + alloc(128 * 144 * 2));
  f16* WcatT  = (f16*)(ws + alloc(256 * 128 * 2));
  f16* rh1Wt  = (f16*)(ws + alloc(128 * 128 * 2));
  f16* rh2Wt  = (f16*)(ws + alloc(9 * 128 * 64 * 2));
  float* vb    = (float*)(ws + alloc(128 * 4));
  float* ba    = (float*)(ws + alloc(128 * 4));
  float* b2    = (float*)(ws + alloc(128 * 4));
  float* soawB = (float*)(ws + alloc(144 * 4));
  float* exW3  = (float*)(ws + alloc(3 * 128 * 4));
  float* wsig  = (float*)(ws + alloc(2 * 128 * 4));
  float* sc1   = (float*)(ws + alloc(B_ * 128 * 4));
  float* sh1   = (float*)(ws + alloc(B_ * 128 * 4));
  float* sc2   = (float*)(ws + alloc(B_ * 64 * 4));
  float* sh2   = (float*)(ws + alloc(B_ * 64 * 4));

  // ---- prep: geometry + weight folds/conversions ----
  k_geom <<<64, 256, 0, stream>>>(geom);
  k_fold <<<64, 256, 0, stream>>>(vp_w, 128, pv_w, 128, vWt, 128);               // vp*pv
  k_fold <<<64, 256, 0, stream>>>(qs1_w, 130, pq_w, 128, WaT, 128);              // qs1[:,:128]*pq
  k_fold <<<64, 256, 0, stream>>>(po_w, 128, op_w, 128, WcatT + 128 * 128, 128); // po*op
  k_convT<<<64, 256, 0, stream>>>(rh1_w, 131, 128, 128, rh1Wt, 128, 0);
  k_convT<<<64, 256, 0, stream>>>(qs2_w, 128, 128, 128, qs2Wt, 128, 0);
  k_convT<<<48, 256, 0, stream>>>(so_w, 128, 128, 96, soawWt, 144, 0);
  k_convT<<<24, 256, 0, stream>>>(aw_w, 128, 128, 48, soawWt, 144, 96);
  k_convT<<<64, 256, 0, stream>>>(po_w, 128, 128, 128, WcatT, 128, 0);
  k_foldb<<<1, 128, 0, stream>>>(vp_w, 128, pv_b, vp_b, vb);
  k_foldb<<<1, 128, 0, stream>>>(qs1_w, 130, pq_b, qs1_b, ba);
  k_foldb<<<1, 128, 0, stream>>>(po_w, 128, op_b, po_b, b2);
  k_misc <<<2, 256, 0, stream>>>(so_b, aw_b, soawB, rh1_w, exW3, qs1_w, wsig);
  k_rh2  <<<288, 256, 0, stream>>>(rh2_w, rh2Wt);

  // ---- value: v = bev * (vp*pv)^T + vb  (fused, f16 out) ----
  gemm_wmma<128, 128, AK_F32_PLANAR, OK_F16, false, 0>
      <<<dim3(NPIX_BEV / 256, B_), 256, 0, stream>>>(
      bev, vWt, vb, nullptr, nullptr, 0, v, NPIX_BEV, 0, 128);

  // ---- rh1: hpre = x*rh1^T + uvec rank-3 + bias (pre-GN) ----
  gemm_wmma<128, 128, AK_F32_PIX, OK_F32, false, 3>
      <<<dim3(NPIX_RV / 256, B_), 256, 0, stream>>>(
      x_rv, rh1Wt, rh1_b, geom, exW3, 0, hpre, NPIX_RV, 128, 128);
  gn_reduce<<<32, 256, 0, stream>>>(hpre, gn1_w, gn1_b, sc1, sh1, 128, 4, NPIX_RV);
  gn_apply_gelu<<<(B_ * NPIX_RV * 128) / 256, 256, 0, stream>>>(hpre, sc1, sh1, h16, 128, NPIX_RV);

  // ---- conv3x3 wrap 128->64 ----
  conv3_wmma<<<dim3(2, 32, B_), 256, 0, stream>>>(h16, rh2Wt, h2pre);
  gn_reduce<<<32, 256, 0, stream>>>(h2pre, gn2_w, gn2_b, sc2, sh2, 64, 3, NPIX_RV);

  // ---- head: mu/sigma/ref/sig_feat (+ writes d_out mu & sigma regions) ----
  float* out_y  = (float*)d_out;
  float* out_mu = out_y + (size_t)B_ * 128 * NPIX_RV;
  float* out_sg = out_mu + (size_t)B_ * NPIX_RV;
  head_kernel<<<(B_ * NPIX_RV) / 256, 256, 0, stream>>>(
      h2pre, sc2, sh2, rh3_w, rh3_b, geom, L, out_mu, out_sg, refb, sigf);

  // ---- query: h1 = gelu(x*(qs1*pq)^T + sig*Wsig^T + ba); Q = h1*qs2^T ----
  gemm_wmma<128, 128, AK_F32_PIX, OK_F16, true, 2>
      <<<dim3(NPIX_RV / 256, B_), 256, 0, stream>>>(
      x_rv, WaT, ba, sigf, wsig, (long)NPIX_RV * 2, h1, NPIX_RV, 128, 128);
  gemm_wmma<128, 128, AK_F16_PIX, OK_F16, false, 0>
      <<<dim3(NPIX_RV / 256, B_), 256, 0, stream>>>(
      h1, qs2Wt, qs2_b, nullptr, nullptr, 0, cat, NPIX_RV, 128, 256);

  // ---- offsets + attention weights: P = Q*[so|aw]^T + b ----
  gemm_wmma<128, 144, AK_F16_PIX, OK_F32, false, 0>
      <<<dim3(NPIX_RV / 256, B_), 256, 0, stream>>>(
      cat, soawWt, soawB, nullptr, nullptr, 0, Pb, NPIX_RV, 256, 144);

  // ---- MSDA bilinear sampling -> cat[:,128:256] ----
  msda_sample<<<(B_ * NPIX_RV * 8) / 256, 256, 0, stream>>>(Pb, refb, v, cat);

  // ---- final: y = [Q|samp]*[po | po*op]^T + b2, planar f32 out ----
  gemm_wmma<256, 128, AK_F16_PIX, OK_PLANAR, false, 0>
      <<<dim3(NPIX_RV / 256, B_), 256, 0, stream>>>(
      cat, WcatT, b2, nullptr, nullptr, 0, out_y, NPIX_RV, 256, 0);
}
